// CasualSelfAttention_51221779972538
// MI455X (gfx1250) — matrix operand order repaired and checked by reading, not compiled, not run on hardware
//
#include <hip/hip_runtime.h>
#include <hip/hip_bf16.h>

typedef __attribute__((ext_vector_type(16))) _Float16 v16h;
typedef __attribute__((ext_vector_type(8)))  _Float16 v8h;
typedef __attribute__((ext_vector_type(4)))  _Float16 v4h;
typedef __attribute__((ext_vector_type(8)))  float    v8f;

#define B_SZ 4
#define S_SZ 2048
#define E_SZ 768
#define H_SZ 12
#define D_SZ 64
#define N_QKV (3*E_SZ)
#define BS_SZ (B_SZ*S_SZ)

static __device__ __forceinline__ v16h mk16(v8h lo, v8h hi){
  v16h r;
#pragma unroll
  for(int j=0;j<8;++j){ r[j]=lo[j]; r[j+8]=hi[j]; }
  return r;
}
static __device__ __forceinline__ v8f wmma_f16(v16h a, v16h b, v8f c){
  // D = A(16x32 f16) * B(32x16 f16) + C(16x16 f32)
  return __builtin_amdgcn_wmma_f32_16x16x32_f16(false, a, false, b, (short)0, c, false, false);
}

// ---------------------------------------------------------------------------
// Prepass: fp32 -> f16 bulk conversion (done ONCE so GEMM staging is pure
// b128 copies; removes ~264M redundant v_cvt from the compute-bound loops)
// ---------------------------------------------------------------------------
__global__ __launch_bounds__(256)
void cvt_f32_to_f16_kernel(const float* __restrict__ src,
                           _Float16* __restrict__ dst, int n4)
{
  const int stride = gridDim.x * blockDim.x;
  for (int i = blockIdx.x * blockDim.x + threadIdx.x; i < n4; i += stride){
    const float4 v = ((const float4*)src)[i];
    v4h t = { (_Float16)v.x, (_Float16)v.y, (_Float16)v.z, (_Float16)v.w };
    ((v4h*)dst)[i] = t;
  }
}

// ---------------------------------------------------------------------------
// Kernel 1: qkv = x16 @ Wattn16 + b_attn, scatter to Q/K/V f16 [B,H,S,D]
// block tile 128x128, K-step 32, double-buffered LDS; 8 waves: 4(M) x 2(N)
// ---------------------------------------------------------------------------
__global__ __launch_bounds__(256)
void qkv_gemm_kernel(const _Float16* __restrict__ X16, const _Float16* __restrict__ W16,
                     const float* __restrict__ bias,
                     _Float16* __restrict__ Qb, _Float16* __restrict__ Kb,
                     _Float16* __restrict__ Vb)
{
  __shared__ __align__(16) _Float16 As[2][128][40];  // [buf][m][k]
  __shared__ __align__(16) _Float16 Bt[2][128][40];  // [buf][n][k] (transposed)
  const int tid  = threadIdx.x;
  const int lane = tid & 31, wid = tid >> 5;
  const int wm = wid & 3, wn = wid >> 2;
  const int lm = lane & 15, kh = lane >> 4;
  const int m0 = blockIdx.y * 128, n0 = blockIdx.x * 128;

  // fixed per-thread staging coordinates
  const int mA[2] = { (tid      ) >> 2, (tid + 256) >> 2 };
  const int kA[2] = { (tid & 3) * 8,    (tid & 3) * 8    };
  const int kB[2] = { (tid      ) >> 4, (tid + 256) >> 4 };
  const int nB    = (tid & 15) * 8;

  v8f acc[2][4];
#pragma unroll
  for(int i=0;i<2;++i)
#pragma unroll
    for(int j=0;j<4;++j) acc[i][j] = (v8f){0,0,0,0,0,0,0,0};

  v8h areg[2], breg[2];
#pragma unroll
  for(int i=0;i<2;++i){
    areg[i] = *(const v8h*)&X16[(size_t)(m0 + mA[i])*E_SZ + kA[i]];
    breg[i] = *(const v8h*)&W16[(size_t)kB[i]*N_QKV + n0 + nB];
  }
#pragma unroll
  for(int i=0;i<2;++i){
    *(v8h*)&As[0][mA[i]][kA[i]] = areg[i];
#pragma unroll
    for(int j=0;j<8;++j) Bt[0][nB+j][kB[i]] = breg[i][j];
  }
  __syncthreads();

  int p = 0;
  for(int kt=0; kt<E_SZ/32; ++kt){
    if (kt + 1 < E_SZ/32){
      const int k0 = (kt+1)*32;
#pragma unroll
      for(int i=0;i<2;++i){
        areg[i] = *(const v8h*)&X16[(size_t)(m0 + mA[i])*E_SZ + k0 + kA[i]];
        breg[i] = *(const v8h*)&W16[(size_t)(k0 + kB[i])*N_QKV + n0 + nB];
      }
    }
    v16h afr[2], bfr[4];
#pragma unroll
    for(int mi=0;mi<2;++mi){
      int row = wm*32 + mi*16 + lm;
      afr[mi] = mk16(*(const v8h*)&As[p][row][kh*8], *(const v8h*)&As[p][row][16 + kh*8]);
    }
#pragma unroll
    for(int ni=0;ni<4;++ni){
      int col = wn*64 + ni*16 + lm;
      bfr[ni] = mk16(*(const v8h*)&Bt[p][col][kh*8], *(const v8h*)&Bt[p][col][16 + kh*8]);
    }
#pragma unroll
    for(int mi=0;mi<2;++mi)
#pragma unroll
      for(int ni=0;ni<4;++ni)
        acc[mi][ni] = wmma_f16(afr[mi], bfr[ni], acc[mi][ni]);
    if (kt + 1 < E_SZ/32){
#pragma unroll
      for(int i=0;i<2;++i){
        *(v8h*)&As[1-p][mA[i]][kA[i]] = areg[i];
#pragma unroll
        for(int j=0;j<8;++j) Bt[1-p][nB+j][kB[i]] = breg[i][j];
      }
    }
    __syncthreads();
    p ^= 1;
  }
  // epilogue: + bias, scatter into Q/K/V [B,H,S,D] as f16
#pragma unroll
  for(int mi=0;mi<2;++mi){
#pragma unroll
    for(int ni=0;ni<4;++ni){
      const int n = n0 + wn*64 + ni*16 + lm;
      const float bv = bias[n];
      const int which = n / E_SZ, e = n % E_SZ;
      const int h = e >> 6, d = e & 63;
      _Float16* dst = (which==0) ? Qb : ((which==1) ? Kb : Vb);
#pragma unroll
      for(int r=0;r<8;++r){
        const int m = m0 + wm*32 + mi*16 + kh*8 + r;
        const int bb = m >> 11, s = m & 2047;
        dst[(((size_t)(bb*H_SZ + h))*S_SZ + s)*D_SZ + d] =
            (_Float16)(acc[mi][ni][r] + bv);
      }
    }
  }
}

// ---------------------------------------------------------------------------
// Kernel 2: flash attention. grid (S/128, B*H). 8 waves * 16 q-rows = 128.
// Double-buffered K/V staging; per-wave P tile round-trip through LDS.
// ---------------------------------------------------------------------------
__global__ __launch_bounds__(256)
void flash_attn_kernel(const _Float16* __restrict__ Qb, const _Float16* __restrict__ Kb,
                       const _Float16* __restrict__ Vb, const int* __restrict__ AM,
                       _Float16* __restrict__ Yb)
{
  __shared__ __align__(16) _Float16 Ks[2][64][72];     // [buf][key][d]
  __shared__ __align__(16) _Float16 Vt[2][64][72];     // [buf][d][key]
  __shared__ __align__(16) _Float16 Ps[8][16][72];     // per-wave P [row][key]
  const int tid  = threadIdx.x;
  const int lane = tid & 31, w = tid >> 5;
  const int lm = lane & 15, kh = lane >> 4;
  const int bh = blockIdx.y, b = bh / H_SZ, h = bh % H_SZ;
  const int q0 = blockIdx.x * 128;
  const size_t head_off = ((size_t)(b*H_SZ + h)) * S_SZ * D_SZ;
  const _Float16* qp = Qb + head_off;
  const _Float16* kp = Kb + head_off;
  const _Float16* vp = Vb + head_off;

  // staging coordinates (fixed): 512 v8h chunks over a 64x64 f16 tile
  const int keyc[2] = { (tid      ) >> 3, (tid + 256) >> 3 };
  const int dc      = (tid & 7) * 8;

  // Q fragments held in registers; D=64 -> 2 WMMA k-steps
  const int qrow_a = q0 + w*16 + lm;
  v16h qf[2];
#pragma unroll
  for(int st=0; st<2; ++st){
    const _Float16* r = &qp[(size_t)qrow_a*D_SZ + st*32 + kh*8];
    qf[st] = mk16(*(const v8h*)r, *(const v8h*)(r + 16));
  }

  float m_i[8], l_i[8];
  v8f O[4];
#pragma unroll
  for(int r=0;r<8;++r){ m_i[r] = -__builtin_inff(); l_i[r] = 0.f; }
#pragma unroll
  for(int di=0;di<4;++di) O[di] = (v8f){0,0,0,0,0,0,0,0};

  const int wmax = q0 + w*16 + 15;
  const int nkt  = q0/64 + 2;          // causal: tiles up to the diagonal

  v8h kreg[2], vreg[2];
#pragma unroll
  for(int i=0;i<2;++i){
    kreg[i] = *(const v8h*)&kp[(size_t)keyc[i]*D_SZ + dc];
    vreg[i] = *(const v8h*)&vp[(size_t)keyc[i]*D_SZ + dc];
  }
#pragma unroll
  for(int i=0;i<2;++i){
    *(v8h*)&Ks[0][keyc[i]][dc] = kreg[i];
#pragma unroll
    for(int j=0;j<8;++j) Vt[0][dc+j][keyc[i]] = vreg[i][j];
  }
  __syncthreads();

  int p = 0;
  for(int kt=0; kt<nkt; ++kt){
    const int k0 = kt*64;
    if (kt + 1 < nkt){
      const int kn = (kt+1)*64;
#pragma unroll
      for(int i=0;i<2;++i){
        kreg[i] = *(const v8h*)&kp[(size_t)(kn + keyc[i])*D_SZ + dc];
        vreg[i] = *(const v8h*)&vp[(size_t)(kn + keyc[i])*D_SZ + dc];
      }
    }
    if (k0 <= wmax){   // wave-uniform branch: EXEC all-ones for WMMA
      v8f sc[4];
#pragma unroll
      for(int ni=0;ni<4;++ni){
        v8f c = (v8f){0,0,0,0,0,0,0,0};
#pragma unroll
        for(int st=0; st<2; ++st){
          const _Float16* kr = &Ks[p][ni*16 + lm][st*32 + kh*8];
          c = wmma_f16(qf[st], mk16(*(const v8h*)kr, *(const v8h*)(kr + 16)), c);
        }
        const int key = k0 + ni*16 + lm;
        const int am  = AM[b*S_SZ + key];
#pragma unroll
        for(int r=0;r<8;++r){
          const int qr = q0 + w*16 + kh*8 + r;
          sc[ni][r] = (key > qr || am == 0) ? -__builtin_inff() : c[r] * 0.125f;
        }
      }
      // online softmax: 8 rows per lane, key columns across 16-lane groups
      float mloc[8], ps[8];
#pragma unroll
      for(int r=0;r<8;++r)
        mloc[r] = fmaxf(fmaxf(sc[0][r], sc[1][r]), fmaxf(sc[2][r], sc[3][r]));
#pragma unroll
      for(int msk=1; msk<16; msk<<=1)
#pragma unroll
        for(int r=0;r<8;++r)
          mloc[r] = fmaxf(mloc[r], __shfl_xor(mloc[r], msk, 16));
#pragma unroll
      for(int r=0;r<8;++r){
        const float mnew = fmaxf(m_i[r], mloc[r]);
        float alpha, p0,p1,p2,p3;
        if (mnew == -__builtin_inff()){            // fully-masked row so far
          alpha = 1.f; p0 = p1 = p2 = p3 = 0.f;
        } else {
          alpha = __expf(m_i[r] - mnew);
          p0 = __expf(sc[0][r] - mnew); p1 = __expf(sc[1][r] - mnew);
          p2 = __expf(sc[2][r] - mnew); p3 = __expf(sc[3][r] - mnew);
        }
        sc[0][r]=p0; sc[1][r]=p1; sc[2][r]=p2; sc[3][r]=p3;
        ps[r]  = p0 + p1 + p2 + p3;
        m_i[r] = mnew;
        l_i[r] *= alpha;
#pragma unroll
        for(int di=0;di<4;++di) O[di][r] *= alpha;
      }
#pragma unroll
      for(int msk=1; msk<16; msk<<=1)
#pragma unroll
        for(int r=0;r<8;++r)
          ps[r] += __shfl_xor(ps[r], msk, 16);
#pragma unroll
      for(int r=0;r<8;++r) l_i[r] += ps[r];
      // P -> LDS (C-layout to A-layout transpose inside the wave)
#pragma unroll
      for(int ni=0;ni<4;++ni)
#pragma unroll
        for(int r=0;r<8;++r)
          Ps[w][kh*8 + r][ni*16 + lm] = (_Float16)sc[ni][r];
      // O += P @ V (contraction over 64 keys -> 2 WMMA k-steps)
#pragma unroll
      for(int st=0; st<2; ++st){
        const _Float16* pr = &Ps[w][lm][st*32 + kh*8];
        v16h pa = mk16(*(const v8h*)pr, *(const v8h*)(pr + 16));
#pragma unroll
        for(int di=0;di<4;++di){
          const _Float16* vr = &Vt[p][di*16 + lm][st*32 + kh*8];
          O[di] = wmma_f16(pa, mk16(*(const v8h*)vr, *(const v8h*)(vr + 16)), O[di]);
        }
      }
    }
    if (kt + 1 < nkt){
#pragma unroll
      for(int i=0;i<2;++i){
        *(v8h*)&Ks[1-p][keyc[i]][dc] = kreg[i];
#pragma unroll
        for(int j=0;j<8;++j) Vt[1-p][dc+j][keyc[i]] = vreg[i][j];
      }
    }
    __syncthreads();
    p ^= 1;
  }
  // normalize and emit Y as f16 [B,S,E]
#pragma unroll
  for(int r=0;r<8;++r){
    const float inv = (l_i[r] > 0.f) ? 1.f / l_i[r] : 0.f;
    const int qr = q0 + w*16 + kh*8 + r;
#pragma unroll
    for(int di=0;di<4;++di){
      const int d = di*16 + lm;
      Yb[((size_t)(b*S_SZ) + qr)*E_SZ + h*D_SZ + d] = (_Float16)(O[di][r] * inv);
    }
  }
}

// ---------------------------------------------------------------------------
// Kernel 3: out = Y @ Wproj16 + b_proj (fp32 out), same double-buffered tiling
// ---------------------------------------------------------------------------
__global__ __launch_bounds__(256)
void proj_gemm_kernel(const _Float16* __restrict__ Y, const _Float16* __restrict__ W16,
                      const float* __restrict__ bias, float* __restrict__ Out)
{
  __shared__ __align__(16) _Float16 As[2][128][40];
  __shared__ __align__(16) _Float16 Bt[2][128][40];
  const int tid  = threadIdx.x;
  const int lane = tid & 31, wid = tid >> 5;
  const int wm = wid & 3, wn = wid >> 2;
  const int lm = lane & 15, kh = lane >> 4;
  const int m0 = blockIdx.y * 128, n0 = blockIdx.x * 128;

  const int mA[2] = { (tid      ) >> 2, (tid + 256) >> 2 };
  const int kA[2] = { (tid & 3) * 8,    (tid & 3) * 8    };
  const int kB[2] = { (tid      ) >> 4, (tid + 256) >> 4 };
  const int nB    = (tid & 15) * 8;

  v8f acc[2][4];
#pragma unroll
  for(int i=0;i<2;++i)
#pragma unroll
    for(int j=0;j<4;++j) acc[i][j] = (v8f){0,0,0,0,0,0,0,0};

  v8h areg[2], breg[2];
#pragma unroll
  for(int i=0;i<2;++i){
    areg[i] = *(const v8h*)&Y[(size_t)(m0 + mA[i])*E_SZ + kA[i]];
    breg[i] = *(const v8h*)&W16[(size_t)kB[i]*E_SZ + n0 + nB];
  }
#pragma unroll
  for(int i=0;i<2;++i){
    *(v8h*)&As[0][mA[i]][kA[i]] = areg[i];
#pragma unroll
    for(int j=0;j<8;++j) Bt[0][nB+j][kB[i]] = breg[i][j];
  }
  __syncthreads();

  int p = 0;
  for(int kt=0; kt<E_SZ/32; ++kt){
    if (kt + 1 < E_SZ/32){
      const int k0 = (kt+1)*32;
#pragma unroll
      for(int i=0;i<2;++i){
        areg[i] = *(const v8h*)&Y[(size_t)(m0 + mA[i])*E_SZ + k0 + kA[i]];
        breg[i] = *(const v8h*)&W16[(size_t)(k0 + kB[i])*E_SZ + n0 + nB];
      }
    }
    v16h afr[2], bfr[4];
#pragma unroll
    for(int mi=0;mi<2;++mi){
      int row = wm*32 + mi*16 + lm;
      afr[mi] = mk16(*(const v8h*)&As[p][row][kh*8], *(const v8h*)&As[p][row][16 + kh*8]);
    }
#pragma unroll
    for(int ni=0;ni<4;++ni){
      int col = wn*64 + ni*16 + lm;
      bfr[ni] = mk16(*(const v8h*)&Bt[p][col][kh*8], *(const v8h*)&Bt[p][col][16 + kh*8]);
    }
#pragma unroll
    for(int mi=0;mi<2;++mi)
#pragma unroll
      for(int ni=0;ni<4;++ni)
        acc[mi][ni] = wmma_f16(afr[mi], bfr[ni], acc[mi][ni]);
    if (kt + 1 < E_SZ/32){
#pragma unroll
      for(int i=0;i<2;++i){
        *(v8h*)&As[1-p][mA[i]][kA[i]] = areg[i];
#pragma unroll
        for(int j=0;j<8;++j) Bt[1-p][nB+j][kB[i]] = breg[i][j];
      }
    }
    __syncthreads();
    p ^= 1;
  }
#pragma unroll
  for(int mi=0;mi<2;++mi){
#pragma unroll
    for(int ni=0;ni<4;++ni){
      const int n = n0 + wn*64 + ni*16 + lm;
      const float bv = bias[n];
#pragma unroll
      for(int r=0;r<8;++r){
        const int m = m0 + wm*32 + mi*16 + kh*8 + r;
        Out[(size_t)m*E_SZ + n] = acc[mi][ni][r] + bv;
      }
    }
  }
}

// ---------------------------------------------------------------------------
extern "C" void kernel_launch(void* const* d_in, const int* in_sizes, int n_in,
                              void* d_out, int out_size, void* d_ws, size_t ws_size,
                              hipStream_t stream)
{
  (void)in_sizes; (void)n_in; (void)out_size; (void)ws_size;
  const float* x        = (const float*)d_in[0];
  const float* W_attn   = (const float*)d_in[1];
  const float* b_attn   = (const float*)d_in[2];
  const float* W_proj   = (const float*)d_in[3];
  const float* b_proj   = (const float*)d_in[4];
  const int*   att_mask = (const int*)d_in[5];
  float* out = (float*)d_out;

  const size_t HB = (size_t)B_SZ * H_SZ * S_SZ * D_SZ;   // 6,291,456 elems
  const size_t XN = (size_t)BS_SZ * E_SZ;                // 6,291,456
  const size_t WAN = (size_t)E_SZ * N_QKV;               // 1,769,472
  const size_t WPN = (size_t)E_SZ * E_SZ;                //   589,824
  _Float16* ws  = (_Float16*)d_ws;
  _Float16* Qb  = ws;
  _Float16* Kb  = Qb  + HB;
  _Float16* Vb  = Kb  + HB;
  _Float16* Yb  = Vb  + HB;
  _Float16* X16 = Yb  + HB;
  _Float16* WA16= X16 + XN;
  _Float16* WP16= WA16+ WAN;

  cvt_f32_to_f16_kernel<<<1024, 256, 0, stream>>>(x,      X16,  (int)(XN  >> 2));
  cvt_f32_to_f16_kernel<<<512,  256, 0, stream>>>(W_attn, WA16, (int)(WAN >> 2));
  cvt_f32_to_f16_kernel<<<256,  256, 0, stream>>>(W_proj, WP16, (int)(WPN >> 2));

  qkv_gemm_kernel<<<dim3(N_QKV/128, BS_SZ/128), 256, 0, stream>>>(
      X16, WA16, b_attn, Qb, Kb, Vb);
  flash_attn_kernel<<<dim3(S_SZ/128, B_SZ*H_SZ), 256, 0, stream>>>(
      Qb, Kb, Vb, att_mask, Yb);
  proj_gemm_kernel<<<dim3(E_SZ/128, BS_SZ/128), 256, 0, stream>>>(
      Yb, WP16, b_proj, out);
}
